// BruteForceMoELinear_24893630447779
// MI455X (gfx1250) — matrix operand order repaired and checked by reading, not compiled, not run on hardware
//
#include <hip/hip_runtime.h>

// ---------------------------------------------------------------------------
// MoE routed linear for MI455X (gfx1250, wave32, WMMA).
// bf16x3 split-precision GEMM (hi*hi + hi*lo + lo*hi, fp32 accumulate).
// ---------------------------------------------------------------------------

constexpr int kExperts = 16;
constexpr int kInFeat  = 1024;
constexpr int kOutFeat = 1024;
constexpr int kBatch   = 8192;

constexpr int MT   = 64;   // token rows per tile
constexpr int NT   = 128;  // output cols per tile
constexpr int KT   = 64;   // K chunk staged in LDS
constexpr int LDSP = 72;   // padded LDS row stride (bf16 elems) -> no bank conflicts
constexpr int kMTiles  = 144;          // ceil((8192 + 16*63)/64)
constexpr int kPermCap = kMTiles * MT; // 9216

// workspace layout (ints): [0..15] counts, [16..32] seg_start,
// [40..55] scatter cursors, [64..64+kPermCap) padded permutation
constexpr int WS_COUNTS = 0;
constexpr int WS_SEG    = 16;
constexpr int WS_CUR    = 40;
constexpr int WS_PERM   = 64;
constexpr int WS_INTS   = WS_PERM + kPermCap;

typedef __attribute__((ext_vector_type(16))) __bf16 v16bf;
typedef __attribute__((ext_vector_type(8)))  float  v8f;

union Frag {
  v16bf v;
  uint4 q[2];
  unsigned short s[16];
};

__device__ __forceinline__ unsigned short f32_to_bf16_rne(float x) {
  unsigned int u = __float_as_uint(x);
  u += 0x7fffu + ((u >> 16) & 1u);
  return (unsigned short)(u >> 16);
}
__device__ __forceinline__ float bf16_to_f32(unsigned short h) {
  return __uint_as_float(((unsigned int)h) << 16);
}

// load 16 contiguous fp32, split into bf16 hi + bf16 lo (residual)
__device__ __forceinline__ void cvt16(const float* __restrict__ src,
                                      Frag& h, Frag& l) {
  float buf[16];
#pragma unroll
  for (int i = 0; i < 4; ++i) ((float4*)buf)[i] = ((const float4*)src)[i];
#pragma unroll
  for (int i = 0; i < 16; ++i) {
    unsigned short hb = f32_to_bf16_rne(buf[i]);
    h.s[i] = hb;
    l.s[i] = f32_to_bf16_rne(buf[i] - bf16_to_f32(hb));
  }
}

// ---------------------------------------------------------------------------
// Prep kernels: counts -> aligned segment starts -> scatter permutation
// ---------------------------------------------------------------------------
__global__ void moe_init(int* __restrict__ ws) {
  int i = blockIdx.x * blockDim.x + threadIdx.x;
  if (i < WS_PERM) {
    if (i < WS_INTS) ws[i] = 0;
  } else if (i < WS_INTS) {
    ws[i] = -1;  // padded permutation slots
  }
}

__global__ void moe_count(const int* __restrict__ gate, int* __restrict__ ws) {
  int t = blockIdx.x * blockDim.x + threadIdx.x;
  if (t < kBatch) atomicAdd(&ws[WS_COUNTS + gate[t]], 1);
}

__global__ void moe_scan(int* __restrict__ ws) {
  if (threadIdx.x == 0 && blockIdx.x == 0) {
    int acc = 0;
    for (int e = 0; e < kExperts; ++e) {
      ws[WS_SEG + e] = acc;
      acc += (ws[WS_COUNTS + e] + MT - 1) / MT * MT;  // 64-align each segment
    }
    ws[WS_SEG + kExperts] = acc;
  }
}

__global__ void moe_scatter(const int* __restrict__ gate, int* __restrict__ ws) {
  int t = blockIdx.x * blockDim.x + threadIdx.x;
  if (t < kBatch) {
    int e = gate[t];
    int p = atomicAdd(&ws[WS_CUR + e], 1);
    ws[WS_PERM + ws[WS_SEG + e] + p] = t;
  }
}

// ---------------------------------------------------------------------------
// Grouped GEMM: one (64 tokens x 128 outputs) tile per block, 8 waves.
// ---------------------------------------------------------------------------
__global__ __launch_bounds__(256)
void moe_gemm(const float* __restrict__ inp, const int* __restrict__ gate,
              const float* __restrict__ weight, const int* __restrict__ ws,
              float* __restrict__ out) {
  __shared__ __align__(16) unsigned short sAhi[MT * LDSP];
  __shared__ __align__(16) unsigned short sAlo[MT * LDSP];
  __shared__ __align__(16) unsigned short sBhi[NT * LDSP];
  __shared__ __align__(16) unsigned short sBlo[NT * LDSP];

  const int* perm = ws + WS_PERM;
  const int tid   = threadIdx.x;
  const int mBase = blockIdx.y * MT;
  const int nBase = blockIdx.x * NT;

  // Scatter fills each segment from its 64-aligned start, so valid tokens are
  // a prefix of the tile: perm[mBase] < 0  <=>  tile entirely empty.
  const int head = perm[mBase];
  if (head < 0) return;
  const int expert = gate[head];
  const float* W = weight + (size_t)expert * kOutFeat * kInFeat;

  const int wave  = tid >> 5;
  const int lane  = tid & 31;
  const int half  = lane >> 4;   // lanes 16-31 hold the other K half
  const int lr    = lane & 15;
  const int m_sub = wave & 3;          // 16-row sub-tile of this wave
  const int nsub0 = (wave >> 2) * 4;   // first of 4 16-col sub-tiles

  // staging assignments
  const int a_row = tid >> 2;            // 0..63
  const int a_col = (tid & 3) * 16;      // 0,16,32,48
  const int a_tok = perm[mBase + a_row];
  const int b_row = tid >> 1;            // 0..127
  const int b_col = (tid & 1) * 32;      // 0,32
  const float* Wrow = W + (size_t)(nBase + b_row) * kInFeat;

  v8f acc[4] = {};

  for (int ko = 0; ko < kInFeat / KT; ++ko) {
    __syncthreads();
    // ---- stage A (gathered token rows, fp32 -> bf16 hi/lo) ----
    {
      Frag h, l;
      if (a_tok >= 0) {
        cvt16(inp + (size_t)a_tok * kInFeat + ko * KT + a_col, h, l);
      } else {
#pragma unroll
        for (int i = 0; i < 16; ++i) { h.s[i] = 0; l.s[i] = 0; }
      }
      unsigned short* dh = &sAhi[a_row * LDSP + a_col];
      unsigned short* dl = &sAlo[a_row * LDSP + a_col];
      *(uint4*)dh = h.q[0]; *(uint4*)(dh + 8) = h.q[1];
      *(uint4*)dl = l.q[0]; *(uint4*)(dl + 8) = l.q[1];
    }
    // ---- stage B (expert weight rows) ----
    {
      const float* src = Wrow + ko * KT + b_col;
      Frag h0, l0, h1, l1;
      cvt16(src, h0, l0);
      cvt16(src + 16, h1, l1);
      unsigned short* dh = &sBhi[b_row * LDSP + b_col];
      unsigned short* dl = &sBlo[b_row * LDSP + b_col];
      *(uint4*)(dh)      = h0.q[0]; *(uint4*)(dh + 8)  = h0.q[1];
      *(uint4*)(dh + 16) = h1.q[0]; *(uint4*)(dh + 24) = h1.q[1];
      *(uint4*)(dl)      = l0.q[0]; *(uint4*)(dl + 8)  = l0.q[1];
      *(uint4*)(dl + 16) = l1.q[0]; *(uint4*)(dl + 24) = l1.q[1];
    }
    // prefetch next K chunk while we compute this one
    if (ko + 1 < kInFeat / KT) {
      __builtin_prefetch(Wrow + (ko + 1) * KT + b_col, 0, 3);
      if (a_tok >= 0)
        __builtin_prefetch(inp + (size_t)a_tok * kInFeat + (ko + 1) * KT + a_col, 0, 3);
    }
    __syncthreads();

    // ---- compute: 2 x (K=32) steps, 4 N sub-tiles, 3 WMMAs each ----
#pragma unroll
    for (int k0 = 0; k0 < KT; k0 += 32) {
      Frag ah, al;
      // A k order used here (same as B): lanes 0-15 hold K[0..15], lanes 16-31
      // hold K[16..31] of row M=lane&15.
      const unsigned short* abh = &sAhi[(m_sub * 16 + lr) * LDSP + k0];
      const unsigned short* abl = &sAlo[(m_sub * 16 + lr) * LDSP + k0];
      ah.q[0] = *(const uint4*)(abh + half * 16);
      ah.q[1] = *(const uint4*)(abh + half * 16 + 8);
      al.q[0] = *(const uint4*)(abl + half * 16);
      al.q[1] = *(const uint4*)(abl + half * 16 + 8);
#pragma unroll
      for (int j = 0; j < 4; ++j) {
        Frag bh, bl;
        // B k order used here (same as A): lane holds column N=lane&15, K contiguous
        // [0..15] (lanes 0-15) or [16..31] (lanes 16-31).
        const unsigned short* bbh =
            &sBhi[((nsub0 + j) * 16 + lr) * LDSP + k0 + half * 16];
        const unsigned short* bbl =
            &sBlo[((nsub0 + j) * 16 + lr) * LDSP + k0 + half * 16];
        bh.q[0] = *(const uint4*)(bbh);
        bh.q[1] = *(const uint4*)(bbh + 8);
        bl.q[0] = *(const uint4*)(bbl);
        bl.q[1] = *(const uint4*)(bbl + 8);
        acc[j] = __builtin_amdgcn_wmma_f32_16x16x32_bf16(
            false, ah.v, false, bh.v, (short)0, acc[j], false, false);
        acc[j] = __builtin_amdgcn_wmma_f32_16x16x32_bf16(
            false, ah.v, false, bl.v, (short)0, acc[j], false, false);
        acc[j] = __builtin_amdgcn_wmma_f32_16x16x32_bf16(
            false, al.v, false, bh.v, (short)0, acc[j], false, false);
      }
    }
  }

  // ---- epilogue: C layout VGPR r -> (M = r + half*8, N = lane&15) ----
  int tok[8];
#pragma unroll
  for (int r = 0; r < 8; ++r)
    tok[r] = perm[mBase + m_sub * 16 + half * 8 + r];
#pragma unroll
  for (int j = 0; j < 4; ++j) {
    const int col = nBase + (nsub0 + j) * 16 + lr;
#pragma unroll
    for (int r = 0; r < 8; ++r) {
      if (tok[r] >= 0) out[(size_t)tok[r] * kOutFeat + col] = acc[j][r];
    }
  }
}

// ---------------------------------------------------------------------------
extern "C" void kernel_launch(void* const* d_in, const int* in_sizes, int n_in,
                              void* d_out, int out_size, void* d_ws, size_t ws_size,
                              hipStream_t stream) {
  const float* inp    = (const float*)d_in[0];
  const int*   gate   = (const int*)d_in[1];
  const float* weight = (const float*)d_in[2];
  float*       out    = (float*)d_out;
  int*         ws     = (int*)d_ws;

  moe_init<<<(WS_INTS + 255) / 256, 256, 0, stream>>>(ws);
  moe_count<<<(kBatch + 255) / 256, 256, 0, stream>>>(gate, ws);
  moe_scan<<<1, 32, 0, stream>>>(ws);
  moe_scatter<<<(kBatch + 255) / 256, 256, 0, stream>>>(gate, ws);

  dim3 grid(kOutFeat / NT, kMTiles);
  moe_gemm<<<grid, 256, 0, stream>>>(inp, gate, weight, ws, out);
}
